// SlotMamba_2121713844506
// MI455X (gfx1250) — matrix operand order repaired and checked by reading, not compiled, not run on hardware
//
#include <hip/hip_runtime.h>
#include <hip/hip_bf16.h>

// ---- problem constants --------------------------------------------------
#define NB      64
#define NK      256
#define DM      512          // d_model
#define DI      512          // d_inner
#define DS      16           // d_state
#define DTR     32           // dt_rank
#define NTOK    (NB * NK)    // 16384 tokens
#define EPSV    1e-5f

typedef __attribute__((ext_vector_type(16))) _Float16 v16h;
typedef __attribute__((ext_vector_type(8)))  float    v8f;

// ---- wave32 reduction ---------------------------------------------------
__device__ __forceinline__ float wave_sum32(float v) {
#pragma unroll
  for (int o = 16; o >= 1; o >>= 1) v += __shfl_xor(v, o, 32);
  return v;
}

// ---- generic f32 -> f16 cast -------------------------------------------
__global__ __launch_bounds__(256) void cast_f32_f16_kernel(
    const float* __restrict__ src, _Float16* __restrict__ dst, int n) {
  int i = blockIdx.x * 256 + threadIdx.x;
  if (i < n) dst[i] = (_Float16)src[i];
}

// ---- LayerNorm (512 wide) + cast to f16; one wave per token -------------
__global__ __launch_bounds__(256) void ln_cast_kernel(
    const float* __restrict__ x, const float* __restrict__ g,
    const float* __restrict__ b, _Float16* __restrict__ out, int ntok) {
  int lane  = threadIdx.x & 31;
  int token = blockIdx.x * 8 + (threadIdx.x >> 5);
  if (token >= ntok) return;
  const float* row = x + (size_t)token * DM;
  float v[16];
  float s = 0.f;
#pragma unroll
  for (int i = 0; i < 4; ++i) {
    float4 f4 = ((const float4*)row)[lane * 4 + i];
    v[i * 4 + 0] = f4.x; v[i * 4 + 1] = f4.y;
    v[i * 4 + 2] = f4.z; v[i * 4 + 3] = f4.w;
    s += f4.x + f4.y + f4.z + f4.w;
  }
  float mean = wave_sum32(s) * (1.f / DM);
  float sq = 0.f;
#pragma unroll
  for (int i = 0; i < 16; ++i) { float d = v[i] - mean; sq += d * d; }
  float rstd = rsqrtf(wave_sum32(sq) * (1.f / DM) + EPSV);
  _Float16* orow = out + (size_t)token * DM;
#pragma unroll
  for (int i = 0; i < 16; ++i) {
    int c = lane * 16 + i;
    orow[c] = (_Float16)((v[i] - mean) * rstd * g[c] + b[c]);
  }
}

// ---- register-blocked WMMA GEMM: C[M,N] = A[M,Kd] * W[N,Kd]^T -----------
// One wave computes a 32x64 output tile (2 M-subtiles x 4 N-subtiles = 8
// v_wmma per K-step). A register images reused across the 4 N-subtiles, B
// across the 2 M-subtiles -> 12 b128 loads per 8 WMMAs. Register images
// follow the CDNA5 wave32 16-bit A(16x32)/B(32x16) layouts (CDNA5 ISA).
__global__ __launch_bounds__(256) void wmma_gemm_f16_kernel(
    const _Float16* __restrict__ A, int lda,
    const _Float16* __restrict__ W, int ldb,
    float* __restrict__ C, int ldc, int M, int N, int Kd) {
  int lane = threadIdx.x & 31;
  int wave = threadIdx.x >> 5;
  int tile = blockIdx.x * 8 + wave;
  int tiles_n = N >> 6;                 // 64-wide N blocks
  int total   = (M >> 5) * tiles_n;     // 32-tall M blocks
  if (tile >= total) return;
  int m0 = (tile / tiles_n) << 5;
  int n0 = (tile % tiles_n) << 6;
  int l15  = lane & 15;
  int half = lane >> 4;

  v8f acc[2][4];
#pragma unroll
  for (int i = 0; i < 2; ++i)
#pragma unroll
    for (int j = 0; j < 4; ++j) acc[i][j] = (v8f){};

  // lane-private row bases (dword aligned: lda/ldb even, offsets even)
  const _Float16* arow0 = A + (size_t)(m0 + l15) * lda + 8 * half;
  const _Float16* arow1 = arow0 + (size_t)16 * lda;
  const _Float16* brow  = W + (size_t)(n0 + l15) * ldb + 8 * half;

  for (int k0 = 0; k0 < Kd; k0 += 32) {
    union { v16h v; unsigned u[8]; } a0, a1, b;
    // A: VGPR0-3 hold K = {0..7}+8*half ; VGPR4-7 hold K = {16..23}+8*half
    const unsigned* ap0 = (const unsigned*)(arow0 + k0);
    const unsigned* ap1 = (const unsigned*)(arow1 + k0);
#pragma unroll
    for (int j = 0; j < 4; ++j) { a0.u[j] = ap0[j]; a0.u[4 + j] = ap0[8 + j]; }
#pragma unroll
    for (int j = 0; j < 4; ++j) { a1.u[j] = ap1[j]; a1.u[4 + j] = ap1[8 + j]; }
#pragma unroll
    for (int nt = 0; nt < 4; ++nt) {
      // B: K = {0..7}+8*half then {16..23}+8*half, N = (n-subtile)*16 + lane&15
      const unsigned* bp =
          (const unsigned*)(brow + (size_t)nt * 16 * ldb + k0);
#pragma unroll
      for (int j = 0; j < 4; ++j) { b.u[j] = bp[j]; b.u[4 + j] = bp[8 + j]; }
      acc[0][nt] = __builtin_amdgcn_wmma_f32_16x16x32_f16(
          false, a0.v, false, b.v, (short)0, acc[0][nt], false, false);
      acc[1][nt] = __builtin_amdgcn_wmma_f32_16x16x32_f16(
          false, a1.v, false, b.v, (short)0, acc[1][nt], false, false);
    }
  }
  // D layout: VGPR r -> row base + 8*half + r, col base + (lane&15)
#pragma unroll
  for (int i = 0; i < 2; ++i) {
#pragma unroll
    for (int nt = 0; nt < 4; ++nt) {
      float* crow =
          C + (size_t)(m0 + 16 * i + 8 * half) * ldc + n0 + 16 * nt + l15;
#pragma unroll
      for (int r = 0; r < 8; ++r) crow[(size_t)r * ldc] = acc[i][nt][r];
    }
  }
}

// ---- causal depthwise conv (win=4) + SiLU; writes f32 and f16 -----------
__global__ __launch_bounds__(256) void conv_silu_kernel(
    const float* __restrict__ xz, const float* __restrict__ cw,
    const float* __restrict__ cb, float* __restrict__ xc,
    _Float16* __restrict__ xc16) {
  int idx = blockIdx.x * 256 + threadIdx.x;     // over NTOK*DI
  if (idx >= NTOK * DI) return;
  int d = idx & (DI - 1);
  int t = idx >> 9;
  int k = t & (NK - 1);
  int b = t >> 8;
  float acc = cb[d];
#pragma unroll
  for (int w = 0; w < 4; ++w) {
    int kk = k + w - 3;
    if (kk >= 0)
      acc += xz[((size_t)(b * NK + kk)) * (2 * DI) + d] * cw[d * 4 + w];
  }
  float s = acc / (1.f + __expf(-acc));         // SiLU
  xc[idx]   = s;
  xc16[idx] = (_Float16)s;
}

// ---- selective scan: one block per batch, one lane per channel ----------
__global__ __launch_bounds__(512) void scan_kernel(
    const float* __restrict__ proj,   // [NTOK,64]: dtr|Bc|Cc
    const float* __restrict__ dtraw,  // [NTOK,DI]
    const float* __restrict__ dtb,    // [DI]
    const float* __restrict__ xc,     // [NTOK,DI]
    const float* __restrict__ xz,     // [NTOK,2*DI] (z = cols DI..)
    const float* __restrict__ A_log,  // [DI,DS]
    const float* __restrict__ Dp,     // [DI]
    _Float16* __restrict__ yg16) {    // [NTOK,DI] gated output (f16)
  int d = threadIdx.x;                // channel
  int b = blockIdx.x;                 // batch
  __shared__ float sBC[8][32];        // 8 steps x (B[16]|C[16])
  float h[DS], Av[DS];
#pragma unroll
  for (int s = 0; s < DS; ++s) {
    h[s]  = 0.f;
    Av[s] = -__expf(A_log[d * DS + s]);
  }
  float bias = dtb[d];
  float Dd   = Dp[d];
  for (int kc = 0; kc < NK; kc += 8) {
    __syncthreads();
    if (threadIdx.x < 256) {
      int kk = threadIdx.x >> 5, c = threadIdx.x & 31;
      sBC[kk][c] = proj[((size_t)(b * NK + kc + kk)) * 64 + DTR + c];
    }
    __syncthreads();
#pragma unroll 1
    for (int kk = 0; kk < 8; ++kk) {
      size_t t = (size_t)(b * NK + kc + kk);
      float dv = dtraw[t * DI + d] + bias;
      dv = (dv > 20.f) ? dv : log1pf(__expf(dv));     // softplus
      float xv = xc[t * DI + d];
      float dx = dv * xv;
      float y  = 0.f;
#pragma unroll
      for (int s = 0; s < DS; ++s) {
        float dA = __expf(dv * Av[s]);
        h[s] = fmaf(dA, h[s], dx * sBC[kk][s]);
        y    = fmaf(h[s], sBC[kk][DS + s], y);
      }
      y = fmaf(xv, Dd, y);
      float zv   = xz[t * (2 * DI) + DI + d];
      float gate = zv / (1.f + __expf(-zv));          // SiLU(z)
      yg16[t * DI + d] = (_Float16)(y * gate);
    }
  }
}

// ---- residual add + final LayerNorm -> f32 out --------------------------
__global__ __launch_bounds__(256) void final_ln_kernel(
    const float* __restrict__ mm2, const float* __restrict__ resid,
    const float* __restrict__ g, const float* __restrict__ b,
    float* __restrict__ out, int ntok) {
  int lane  = threadIdx.x & 31;
  int token = blockIdx.x * 8 + (threadIdx.x >> 5);
  if (token >= ntok) return;
  const float* r1 = mm2   + (size_t)token * DM;
  const float* r2 = resid + (size_t)token * DM;
  float v[16];
  float s = 0.f;
#pragma unroll
  for (int i = 0; i < 4; ++i) {
    float4 a4 = ((const float4*)r1)[lane * 4 + i];
    float4 b4 = ((const float4*)r2)[lane * 4 + i];
    v[i * 4 + 0] = a4.x + b4.x; v[i * 4 + 1] = a4.y + b4.y;
    v[i * 4 + 2] = a4.z + b4.z; v[i * 4 + 3] = a4.w + b4.w;
    s += v[i * 4 + 0] + v[i * 4 + 1] + v[i * 4 + 2] + v[i * 4 + 3];
  }
  float mean = wave_sum32(s) * (1.f / DM);
  float sq = 0.f;
#pragma unroll
  for (int i = 0; i < 16; ++i) { float d = v[i] - mean; sq += d * d; }
  float rstd = rsqrtf(wave_sum32(sq) * (1.f / DM) + EPSV);
  float* orow = out + (size_t)token * DM;
#pragma unroll
  for (int i = 0; i < 16; ++i) {
    int c = lane * 16 + i;
    orow[c] = (v[i] - mean) * rstd * g[c] + b[c];
  }
}

// ---- host launcher ------------------------------------------------------
extern "C" void kernel_launch(void* const* d_in, const int* in_sizes, int n_in,
                              void* d_out, int out_size, void* d_ws,
                              size_t ws_size, hipStream_t stream) {
  (void)in_sizes; (void)n_in; (void)out_size; (void)ws_size;
  const float* slots     = (const float*)d_in[0];
  const float* ln_g      = (const float*)d_in[1];
  const float* ln_b      = (const float*)d_in[2];
  const float* in_proj_w = (const float*)d_in[3];   // [1024,512]
  const float* conv_w    = (const float*)d_in[4];   // [512,4]
  const float* conv_b    = (const float*)d_in[5];
  const float* x_proj_w  = (const float*)d_in[6];   // [64,512]
  const float* dt_proj_w = (const float*)d_in[7];   // [512,32]
  const float* dt_proj_b = (const float*)d_in[8];
  const float* A_log     = (const float*)d_in[9];   // [512,16]
  const float* Dp        = (const float*)d_in[10];
  const float* out_proj_w= (const float*)d_in[11];  // [512,512]
  const float* fln_g     = (const float*)d_in[12];
  const float* fln_b     = (const float*)d_in[13];
  float* out = (float*)d_out;

  char* ws = (char*)d_ws;
  size_t off = 0;
  auto take = [&](size_t bytes) -> size_t {
    size_t o = off; off += (bytes + 255) & ~(size_t)255; return o;
  };
  _Float16* xn16   = (_Float16*)(ws + take((size_t)NTOK * DM * 2));
  _Float16* w_in16 = (_Float16*)(ws + take((size_t)2 * DI * DM * 2));
  _Float16* w_x16  = (_Float16*)(ws + take((size_t)(DTR + 2 * DS) * DI * 2));
  _Float16* w_dt16 = (_Float16*)(ws + take((size_t)DI * DTR * 2));
  _Float16* w_out16= (_Float16*)(ws + take((size_t)DM * DI * 2));
  float*    xz     = (float*)   (ws + take((size_t)NTOK * 2 * DI * 4));
  float*    xc     = (float*)   (ws + take((size_t)NTOK * DI * 4));
  _Float16* xc16   = (_Float16*)(ws + take((size_t)NTOK * DI * 2));
  float*    proj   = (float*)   (ws + take((size_t)NTOK * 64 * 4));
  _Float16* proj16 = (_Float16*)(ws + take((size_t)NTOK * 64 * 2));
  float*    dtraw  = (float*)   (ws + take((size_t)NTOK * DI * 4));
  // aliases (lifetimes do not overlap):
  _Float16* yg16 = xc16;   // xc16 dead after x_proj GEMM
  float*    mm2  = xz;     // xz dead after scan

  // 1) weights -> f16
  cast_f32_f16_kernel<<<(2 * DI * DM + 255) / 256, 256, 0, stream>>>(
      in_proj_w, w_in16, 2 * DI * DM);
  cast_f32_f16_kernel<<<((DTR + 2 * DS) * DI + 255) / 256, 256, 0, stream>>>(
      x_proj_w, w_x16, (DTR + 2 * DS) * DI);
  cast_f32_f16_kernel<<<(DI * DTR + 255) / 256, 256, 0, stream>>>(
      dt_proj_w, w_dt16, DI * DTR);
  cast_f32_f16_kernel<<<(DM * DI + 255) / 256, 256, 0, stream>>>(
      out_proj_w, w_out16, DM * DI);

  // 2) LN + cast
  ln_cast_kernel<<<NTOK / 8, 256, 0, stream>>>(slots, ln_g, ln_b, xn16, NTOK);

  // 3) in_proj: [16384,512] x [1024,512]^T -> xz [16384,1024]
  {
    int tiles = (NTOK / 32) * (2 * DI / 64);   // 32x64 per wave
    wmma_gemm_f16_kernel<<<tiles / 8, 256, 0, stream>>>(
        xn16, DM, w_in16, DM, xz, 2 * DI, NTOK, 2 * DI, DM);
  }

  // 4) conv + SiLU
  conv_silu_kernel<<<(NTOK * DI) / 256, 256, 0, stream>>>(
      xz, conv_w, conv_b, xc, xc16);

  // 5) x_proj: [16384,512] x [64,512]^T -> proj [16384,64]
  {
    int tiles = (NTOK / 32) * (64 / 64);
    wmma_gemm_f16_kernel<<<tiles / 8, 256, 0, stream>>>(
        xc16, DI, w_x16, DI, proj, 64, NTOK, 64, DI);
  }

  // 6) proj -> f16 (dtr columns used as GEMM A with lda=64)
  cast_f32_f16_kernel<<<(NTOK * 64) / 256, 256, 0, stream>>>(
      proj, proj16, NTOK * 64);

  // 7) dt_proj: [16384,32] x [512,32]^T -> dtraw [16384,512]
  {
    int tiles = (NTOK / 32) * (DI / 64);
    wmma_gemm_f16_kernel<<<tiles / 8, 256, 0, stream>>>(
        proj16, 64, w_dt16, DTR, dtraw, DI, NTOK, DI, DTR);
  }

  // 8) selective scan (+softplus, +D skip, +SiLU(z) gate) -> yg16
  scan_kernel<<<NB, DI, 0, stream>>>(
      proj, dtraw, dt_proj_b, xc, xz, A_log, Dp, yg16);

  // 9) out_proj: [16384,512] x [512,512]^T -> mm2 [16384,512]
  {
    int tiles = (NTOK / 32) * (DM / 64);
    wmma_gemm_f16_kernel<<<tiles / 8, 256, 0, stream>>>(
        yg16, DI, w_out16, DI, mm2, DM, NTOK, DM, DI);
  }

  // 10) residual + final LN -> d_out (f32)
  final_ln_kernel<<<NTOK / 8, 256, 0, stream>>>(
      mm2, slots, fln_g, fln_b, out, NTOK);
}
